// AttnBlock_11587821765020
// MI455X (gfx1250) — matrix operand order repaired and checked by reading, not compiled, not run on hardware
//
#include <hip/hip_runtime.h>

#define Bn 8
#define Cn 512
#define Wn 2048
#define Gn 32
#define CPG 16
#define KT 32   // K-tile for LDS-staged GEMMs

typedef __bf16 bf16;
typedef __attribute__((ext_vector_type(16))) __bf16 v16bf;
typedef __attribute__((ext_vector_type(8)))  __bf16 v8bf;
typedef __attribute__((ext_vector_type(8)))  float  v8f;

__device__ __forceinline__ v8f wmma_bf16(v16bf a, v16bf b, v8f c) {
  return __builtin_amdgcn_wmma_f32_16x16x32_bf16(false, a, false, b, (short)0, c,
                                                 false, false);
}

// A fragment (16x32 bf16 MxK): lane holds row M=lane&15; elems 0..7 = K(half*8..+7),
// elems 8..15 = K(16+half*8..+7). Two contiguous 16B loads.
__device__ __forceinline__ v16bf make_afrag(const bf16* row, int k0, int hf) {
  v8bf lo = *(const v8bf*)(row + k0 + hf * 8);
  v8bf hi = *(const v8bf*)(row + k0 + hf * 8 + 16);
  v16bf f;
#pragma unroll
  for (int i = 0; i < 8; ++i) { f[i] = lo[i]; f[i + 8] = hi[i]; }
  return f;
}

// B fragment (32x16 bf16 KxN): lane holds col N=lane&15; elems 0..7 = K(half*8..+7),
// elems 8..15 = K(16+half*8..+7). Two 16B loads from an N-major (row = fixed n, varying k) buffer.
__device__ __forceinline__ v16bf make_bfrag(const bf16* rowT, int k0, int hf) {
  v8bf lo = *(const v8bf*)(rowT + k0 + hf * 8);
  v8bf hi = *(const v8bf*)(rowT + k0 + hf * 8 + 16);
  v16bf f;
#pragma unroll
  for (int i = 0; i < 8; ++i) { f[i] = lo[i]; f[i + 8] = hi[i]; }
  return f;
}

__device__ __forceinline__ v8f zero8() {
  v8f z = {0.f, 0.f, 0.f, 0.f, 0.f, 0.f, 0.f, 0.f};
  return z;
}

// Low 32 bits of a generic pointer to __shared__ == wave-relative LDS byte offset
__device__ __forceinline__ unsigned lds_off(const void* p) {
  return (unsigned)(uintptr_t)p;
}

// Async copy of a 64-row x 32-bf16 tile (row stride Cn in global, 64B rows in LDS)
// using global_load_async_to_lds_b128 (GVS mode). 2 instructions per thread,
// 128 threads cover 64*64B = 4KB. Tracked by ASYNCcnt.
__device__ __forceinline__ void async_copy_tile(const bf16* gbase, unsigned ldsbase,
                                                int k0, int tid) {
#pragma unroll
  for (int t = 0; t < 2; ++t) {
    int ch = tid + t * 128;        // 0..255
    int row = ch >> 2;             // 0..63
    int p = ch & 3;                // 16B chunk within 64B row
    unsigned voff = (unsigned)((row * Cn + k0) * 2 + p * 16);
    unsigned laddr = ldsbase + (unsigned)(row * (KT * 2) + p * 16);
    asm volatile("global_load_async_to_lds_b128 %0, %1, %2 offset:0"
                 :
                 : "v"(laddr), "v"(voff), "s"(gbase)
                 : "memory");
  }
}

// 64x64 output tile GEMM over K=Cn, double-buffered async LDS staging.
// Block = 128 threads (4 waves); wave w owns rows [w*16, w*16+16), 4 n-subtiles.
// All 5 fragments loaded before the 4 independent WMMAs (single dscnt wait, no
// accumulator PHI copies thanks to named locals).
__device__ __forceinline__ void gemm64x64(const bf16* Abase, const bf16* Bbase,
                                          bf16* ldsA, bf16* ldsB, int tid,
                                          v8f acc[4]) {
  int lane = tid & 31, wv = tid >> 5, hf = lane >> 4, lm = lane & 15;
  unsigned aoff = lds_off(ldsA);
  unsigned boff = lds_off(ldsB);
  v8f a0 = zero8(), a1 = zero8(), a2 = zero8(), a3 = zero8();
  async_copy_tile(Abase, aoff, 0, tid);
  async_copy_tile(Bbase, boff, 0, tid);
  for (int kt = 0; kt < Cn / KT; ++kt) {
    int buf = kt & 1;
    asm volatile("s_wait_asynccnt 0x0" ::: "memory");
    __syncthreads();  // tile `buf` fully resident in LDS for all waves
    if (kt + 1 < Cn / KT) {
      unsigned nb = (unsigned)((buf ^ 1) * (64 * KT * 2));
      async_copy_tile(Abase, aoff + nb, (kt + 1) * KT, tid);
      async_copy_tile(Bbase, boff + nb, (kt + 1) * KT, tid);
    }
    const bf16* bbase = ldsB + buf * (64 * KT);
    v16bf af = make_afrag(ldsA + buf * (64 * KT) + (wv * 16 + lm) * KT, 0, hf);
    v16bf b0 = make_bfrag(bbase + lm * KT, 0, hf);
    v16bf b1 = make_bfrag(bbase + (16 + lm) * KT, 0, hf);
    v16bf b2 = make_bfrag(bbase + (32 + lm) * KT, 0, hf);
    v16bf b3 = make_bfrag(bbase + (48 + lm) * KT, 0, hf);
    a0 = wmma_bf16(af, b0, a0);
    a1 = wmma_bf16(af, b1, a1);
    a2 = wmma_bf16(af, b2, a2);
    a3 = wmma_bf16(af, b3, a3);
    __syncthreads();  // all waves done reading `buf` before it is refilled
  }
  acc[0] = a0; acc[1] = a1; acc[2] = a2; acc[3] = a3;
}

// ---------------- weights fp32 -> bf16 ----------------
__global__ void wconvert_kernel(const float* wq, const float* wk, const float* wv,
                                const float* wp, bf16* wbf) {
  int id = blockIdx.x * 256 + threadIdx.x;
  int sel = id >> 18;  // C*C = 2^18
  const float* src = sel == 0 ? wq : sel == 1 ? wk : sel == 2 ? wv : wp;
  wbf[id] = (bf16)src[id & (Cn * Cn - 1)];
}

// ---------------- GroupNorm: x(b,c,w) -> hT(b,w,c) bf16 ----------------
__global__ void __launch_bounds__(256) gn_kernel(const float* x, const float* gamma,
                                                 const float* beta, bf16* hT) {
  int b = blockIdx.x >> 5;
  int g = blockIdx.x & 31;
  int tid = threadIdx.x;
  const float* xs = x + ((size_t)b * Cn + g * CPG) * Wn;  // contiguous 16x2048 slab
  const int N = CPG * Wn;
  float s = 0.f, q = 0.f;
  for (int i = tid; i < N; i += 256) { float v = xs[i]; s += v; q += v * v; }
  __shared__ float rs[8], rq[8];
  __shared__ float smean, sinv;
  int lane = tid & 31, wv = tid >> 5;
#pragma unroll
  for (int d = 16; d > 0; d >>= 1) { s += __shfl_xor(s, d, 32); q += __shfl_xor(q, d, 32); }
  if (lane == 0) { rs[wv] = s; rq[wv] = q; }
  __syncthreads();
  if (tid == 0) {
    float ts = 0.f, tq = 0.f;
    for (int i = 0; i < 8; ++i) { ts += rs[i]; tq += rq[i]; }
    float mean = ts / (float)N;
    float var = tq / (float)N - mean * mean;
    smean = mean;
    sinv = rsqrtf(var + 1e-6f);
  }
  __syncthreads();
  float mean = smean, inv = sinv;
  int cl = tid & 15;  // constant per thread (stride 256 keeps low bits)
  int c = g * CPG + cl;
  float ga = gamma[c], be = beta[c];
  for (int j = tid; j < N; j += 256) {
    int w = j >> 4;
    float val = (xs[cl * Wn + w] - mean) * inv * ga + be;
    hT[((size_t)b * Wn + w) * Cn + c] = (bf16)val;  // coalesced over c
  }
}

// ---------------- QKV: out[o,w] = sum_c W[o,c]*h[c,w] + bias ----------------
// q,k stored transposed (b,w,c); v stored (b,c,w). Softmax scale folded into q.
__global__ void __launch_bounds__(128) qkv_kernel(const bf16* hT, const bf16* wbf,
                                                  const float* bq, const float* bk,
                                                  const float* bv, bf16* qT, bf16* kT,
                                                  bf16* vb) {
  __shared__ bf16 ldsA[2][64 * KT];
  __shared__ bf16 ldsB[2][64 * KT];
  int tid = threadIdx.x, lane = tid & 31, wv = tid >> 5, hf = lane >> 4, lm = lane & 15;
  int n0 = blockIdx.x * 64, m0 = blockIdx.y * 64;
  int b = blockIdx.z / 3, which = blockIdx.z % 3;
  const float* bias = which == 0 ? bq : which == 1 ? bk : bv;
  float scale = which == 0 ? 0.044194173824159216f : 1.0f;  // 512^-0.5
  int mr = m0 + wv * 16;
  const bf16* Abase = wbf + (size_t)which * Cn * Cn + (size_t)m0 * Cn;
  const bf16* Bbase = hT + ((size_t)b * Wn + n0) * Cn;
  v8f acc[4];
  gemm64x64(Abase, Bbase, &ldsA[0][0], &ldsB[0][0], tid, acc);
  float bias8[8];
#pragma unroll
  for (int r = 0; r < 8; ++r) bias8[r] = bias[mr + hf * 8 + r];
  if (which < 2) {
    bf16* dst = (which == 0 ? qT : kT) + (size_t)b * Wn * Cn;
#pragma unroll
    for (int n = 0; n < 4; ++n) {
      int wpos = n0 + n * 16 + lm;
      bf16* drow = dst + (size_t)wpos * Cn + mr + hf * 8;
#pragma unroll
      for (int r = 0; r < 8; r += 2) {
        union { bf16 h[2]; unsigned u; } pk;
        pk.h[0] = (bf16)(acc[n][r] * scale + bias8[r]);
        pk.h[1] = (bf16)(acc[n][r + 1] * scale + bias8[r + 1]);
        *(unsigned*)(drow + r) = pk.u;
      }
    }
  } else {
    bf16* dst = vb + (size_t)b * Cn * Wn;
#pragma unroll
    for (int n = 0; n < 4; ++n) {
      int wpos = n0 + n * 16 + lm;
#pragma unroll
      for (int r = 0; r < 8; ++r) {
        int o = mr + hf * 8 + r;
        dst[(size_t)o * Wn + wpos] = (bf16)(acc[n][r] + bias8[r]);
      }
    }
  }
}

// ---------------- attention pass 1: row max + sumexp ----------------
__global__ void __launch_bounds__(256) stats_kernel(const bf16* qT, const bf16* kT,
                                                    float* stats) {
  __shared__ bf16 qTile[32 * Cn];  // 32 queries x 512 c (32KB)
  __shared__ float redm[8][32], redl[8][32];
  int tid = threadIdx.x, lane = tid & 31, wv = tid >> 5, hf = lane >> 4, lm = lane & 15;
  int i0 = blockIdx.x * 32, b = blockIdx.y;
  {
    const uint4* src = (const uint4*)(qT + ((size_t)b * Wn + i0) * Cn);
    uint4* d = (uint4*)qTile;
#pragma unroll
    for (int t = 0; t < 8; ++t) d[tid + t * 256] = src[tid + t * 256];
  }
  __syncthreads();
  float mst[16], lst[16];
#pragma unroll
  for (int t = 0; t < 16; ++t) { mst[t] = -1e30f; lst[t] = 0.f; }
  for (int j0 = 0; j0 < Wn; j0 += 128) {
    const bf16* krow = kT + ((size_t)b * Wn + j0 + wv * 16 + lm) * Cn;
    if (j0 + 128 < Wn)
      __builtin_prefetch(krow + (size_t)128 * Cn, 0, 0);  // global_prefetch_b8
    v8f acc0 = zero8(), acc1 = zero8();
    v16bf bfr = make_bfrag(krow, 0, hf);  // software-pipelined global B stream
    for (int k0 = 0; k0 < Cn; k0 += 32) {
      v16bf bcur = bfr;
      if (k0 + 32 < Cn) bfr = make_bfrag(krow, k0 + 32, hf);
      v16bf af0 = make_afrag(qTile + lm * Cn, k0, hf);
      v16bf af1 = make_afrag(qTile + (16 + lm) * Cn, k0, hf);
      acc0 = wmma_bf16(af0, bcur, acc0);
      acc1 = wmma_bf16(af1, bcur, acc1);
    }
#pragma unroll
    for (int sub = 0; sub < 2; ++sub)
#pragma unroll
      for (int r = 0; r < 8; ++r) {
        int t = sub * 8 + r;
        float s = sub == 0 ? acc0[r] : acc1[r];
        float mn = fmaxf(mst[t], s);
        lst[t] = lst[t] * __expf(mst[t] - mn) + __expf(s - mn);
        mst[t] = mn;
      }
  }
#pragma unroll
  for (int d = 1; d < 16; d <<= 1)
#pragma unroll
    for (int t = 0; t < 16; ++t) {
      float mo = __shfl_xor(mst[t], d, 32);
      float lo = __shfl_xor(lst[t], d, 32);
      float mn = fmaxf(mst[t], mo);
      lst[t] = lst[t] * __expf(mst[t] - mn) + lo * __expf(mo - mn);
      mst[t] = mn;
    }
  if (lm == 0) {
#pragma unroll
    for (int sub = 0; sub < 2; ++sub)
#pragma unroll
      for (int r = 0; r < 8; ++r) {
        int il = sub * 16 + hf * 8 + r;
        redm[wv][il] = mst[sub * 8 + r];
        redl[wv][il] = lst[sub * 8 + r];
      }
  }
  __syncthreads();
  if (tid < 32) {
    float m = -1e30f, l = 0.f;
#pragma unroll
    for (int w = 0; w < 8; ++w) {
      float mo = redm[w][tid], lo = redl[w][tid];
      float mn = fmaxf(m, mo);
      l = l * __expf(m - mn) + lo * __expf(mo - mn);
      m = mn;
    }
    size_t idx = ((size_t)b * Wn + i0 + tid) * 2;
    stats[idx] = m;
    stats[idx + 1] = l;
  }
}

// ---------------- attention pass 2: out = V * softmax(S), store (b,w,c) ----------------
__global__ void __launch_bounds__(256) attout_kernel(const bf16* qT, const bf16* kT,
                                                     const bf16* vb, const float* stats,
                                                     bf16* attT) {
  __shared__ bf16 qTile[32 * Cn];   // 32KB
  __shared__ bf16 pTile[32 * 64];   // 4KB: P[i_local][j_local]
  int tid = threadIdx.x, lane = tid & 31, wv = tid >> 5, hf = lane >> 4, lm = lane & 15;
  int i0 = blockIdx.x * 32, b = blockIdx.y;
  int sub_i = wv & 1, sub_j = wv >> 1, c0 = wv * 64;
  {
    const uint4* src = (const uint4*)(qT + ((size_t)b * Wn + i0) * Cn);
    uint4* d = (uint4*)qTile;
#pragma unroll
    for (int t = 0; t < 8; ++t) d[tid + t * 256] = src[tid + t * 256];
  }
  float mrow[8], linv[8];
#pragma unroll
  for (int r = 0; r < 8; ++r) {
    size_t idx = ((size_t)b * Wn + i0 + sub_i * 16 + hf * 8 + r) * 2;
    mrow[r] = stats[idx];
    linv[r] = 1.f / stats[idx + 1];
  }
  __syncthreads();
  v8f accpv[8];
#pragma unroll
  for (int t = 0; t < 8; ++t) accpv[t] = zero8();
  for (int j0 = 0; j0 < Wn; j0 += 64) {
    // each wave computes one 16x16 S subtile (sub_i, sub_j), K=512
    v8f sacc = zero8();
    const bf16* krow = kT + ((size_t)b * Wn + j0 + sub_j * 16 + lm) * Cn;
    if (j0 + 64 < Wn) __builtin_prefetch(krow + (size_t)64 * Cn, 0, 0);
    v16bf bfr = make_bfrag(krow, 0, hf);  // pipelined global B stream
    for (int k0 = 0; k0 < Cn; k0 += 32) {
      v16bf bcur = bfr;
      if (k0 + 32 < Cn) bfr = make_bfrag(krow, k0 + 32, hf);
      v16bf af = make_afrag(qTile + (sub_i * 16 + lm) * Cn, k0, hf);
      sacc = wmma_bf16(af, bcur, sacc);
    }
#pragma unroll
    for (int r = 0; r < 8; ++r) {
      float p = __expf(sacc[r] - mrow[r]) * linv[r];
      pTile[(sub_i * 16 + hf * 8 + r) * 64 + sub_j * 16 + lm] = (bf16)p;
    }
    __syncthreads();
    // PV: out[c,i] += sum_j v[c,j] * P[i,j]; wave owns c-chunk c0..c0+63
    const bf16* vbase = vb + ((size_t)b * Cn + c0) * Wn + j0;
    if (j0 + 64 < Wn) __builtin_prefetch(vbase + (lm * (size_t)Wn) + 64, 0, 0);
#pragma unroll
    for (int kk = 0; kk < 2; ++kk) {
      v16bf bp0 = make_bfrag(pTile + lm * 64, kk * 32, hf);
      v16bf bp1 = make_bfrag(pTile + (16 + lm) * 64, kk * 32, hf);
      v16bf av = make_afrag(vbase + (size_t)lm * Wn, kk * 32, hf);  // mc=0
#pragma unroll
      for (int mc = 0; mc < 4; ++mc) {
        v16bf acur = av;
        if (mc < 3)
          av = make_afrag(vbase + (size_t)((mc + 1) * 16 + lm) * Wn, kk * 32, hf);
        accpv[mc * 2 + 0] = wmma_bf16(acur, bp0, accpv[mc * 2 + 0]);
        accpv[mc * 2 + 1] = wmma_bf16(acur, bp1, accpv[mc * 2 + 1]);
      }
    }
    __syncthreads();
  }
#pragma unroll
  for (int mc = 0; mc < 4; ++mc)
#pragma unroll
    for (int ni = 0; ni < 2; ++ni) {
      int i = i0 + ni * 16 + lm;
      bf16* drow = attT + ((size_t)b * Wn + i) * Cn + c0 + mc * 16 + hf * 8;
      v8f a = accpv[mc * 2 + ni];
#pragma unroll
      for (int r = 0; r < 8; r += 2) {
        union { bf16 h[2]; unsigned u; } pk;
        pk.h[0] = (bf16)a[r];
        pk.h[1] = (bf16)a[r + 1];
        *(unsigned*)(drow + r) = pk.u;
      }
    }
}

// ---------------- proj + residual: out = x + Wp*att + bp (fp32) ----------------
__global__ void __launch_bounds__(128) proj_kernel(const bf16* attT, const bf16* wbf,
                                                   const float* bp, const float* x,
                                                   float* out) {
  __shared__ bf16 ldsA[2][64 * KT];
  __shared__ bf16 ldsB[2][64 * KT];
  int tid = threadIdx.x, lane = tid & 31, wv = tid >> 5, hf = lane >> 4, lm = lane & 15;
  int n0 = blockIdx.x * 64, m0 = blockIdx.y * 64, b = blockIdx.z;
  int mr = m0 + wv * 16;
  const bf16* Abase = wbf + (size_t)3 * Cn * Cn + (size_t)m0 * Cn;
  const bf16* Bbase = attT + ((size_t)b * Wn + n0) * Cn;
  v8f acc[4];
  gemm64x64(Abase, Bbase, &ldsA[0][0], &ldsB[0][0], tid, acc);
  float bias8[8];
#pragma unroll
  for (int r = 0; r < 8; ++r) bias8[r] = bp[mr + hf * 8 + r];
#pragma unroll
  for (int n = 0; n < 4; ++n) {
    int wpos = n0 + n * 16 + lm;
#pragma unroll
    for (int r = 0; r < 8; ++r) {
      int o = mr + hf * 8 + r;
      size_t idx = ((size_t)b * Cn + o) * Wn + wpos;
      out[idx] = x[idx] + acc[n][r] + bias8[r];
    }
  }
}

extern "C" void kernel_launch(void* const* d_in, const int* in_sizes, int n_in,
                              void* d_out, int out_size, void* d_ws, size_t ws_size,
                              hipStream_t stream) {
  const float* x  = (const float*)d_in[0];
  const float* gg = (const float*)d_in[1];
  const float* gb = (const float*)d_in[2];
  const float* wq = (const float*)d_in[3];
  const float* bq = (const float*)d_in[4];
  const float* wk = (const float*)d_in[5];
  const float* bk = (const float*)d_in[6];
  const float* wv = (const float*)d_in[7];
  const float* bv = (const float*)d_in[8];
  const float* wp = (const float*)d_in[9];
  const float* bp = (const float*)d_in[10];
  float* out = (float*)d_out;

  // workspace layout (needs ~83 MB)
  char* ws = (char*)d_ws;
  const size_t MB = (size_t)1 << 20;
  bf16* hT    = (bf16*)(ws + 0 * MB);    // (b,w,c) 16MB
  bf16* qT    = (bf16*)(ws + 16 * MB);   // (b,w,c) 16MB, scale folded
  bf16* kT    = (bf16*)(ws + 32 * MB);   // (b,w,c) 16MB
  bf16* vb    = (bf16*)(ws + 48 * MB);   // (b,c,w) 16MB
  bf16* attT  = (bf16*)(ws + 64 * MB);   // (b,w,c) 16MB
  float* stats = (float*)(ws + 80 * MB); // (b,w,{m,l}) 128KB
  bf16* wbf   = (bf16*)(ws + 81 * MB);   // 4 x 512x512 bf16 weights, 2MB

  wconvert_kernel<<<4 * Cn * Cn / 256, 256, 0, stream>>>(wq, wk, wv, wp, wbf);
  gn_kernel<<<Bn * Gn, 256, 0, stream>>>(x, gg, gb, hT);
  qkv_kernel<<<dim3(Wn / 64, Cn / 64, Bn * 3), 128, 0, stream>>>(hT, wbf, bq, bk, bv,
                                                                 qT, kT, vb);
  stats_kernel<<<dim3(Wn / 32, Bn), 256, 0, stream>>>(qT, kT, stats);
  attout_kernel<<<dim3(Wn / 32, Bn), 256, 0, stream>>>(qT, kT, vb, stats, attT);
  proj_kernel<<<dim3(Wn / 64, Cn / 64, Bn), 128, 0, stream>>>(attT, wbf, bp, x, out);
}
